// Net_84275848282727
// MI455X (gfx1250) — compile-verified, matrix operand order checked by reading, not run on hardware
//
#include <hip/hip_runtime.h>
#include <hip/hip_bf16.h>

#define N_NODES 50000
#define N_EDGES 800000
#define IN_CH   256
#define ROWTILES 3125   // 50000 / 16
#define COLTILES 16     // 256 / 16
#define KTILES   8      // 256 / 32

typedef __bf16 bf16_t;
typedef __attribute__((ext_vector_type(16))) __bf16 v16bf;
typedef __attribute__((ext_vector_type(8)))  __bf16 v8bf;
typedef __attribute__((ext_vector_type(8)))  float  v8f;

// ---------- helpers ----------
__device__ inline unsigned ford(float f) {
    unsigned u = __float_as_uint(f);
    return (u & 0x80000000u) ? ~u : (u | 0x80000000u);
}
__device__ inline float funord(unsigned u) {
    return (u & 0x80000000u) ? __uint_as_float(u ^ 0x80000000u) : __uint_as_float(~u);
}

__global__ void fill_u32(unsigned* __restrict__ p, unsigned v, int n) {
    int i = blockIdx.x * blockDim.x + threadIdx.x;
    if (i < n) p[i] = v;
}

// ---------- BatchNorm ----------
__global__ __launch_bounds__(256) void bn_reduce(const float* __restrict__ x,
                                                 float* __restrict__ sum,
                                                 float* __restrict__ sumsq, int rows) {
    int c = threadIdx.x;  // 256 channels
    float s = 0.f, s2 = 0.f;
    for (int r = blockIdx.x; r < rows; r += gridDim.x) {
        float v = x[(size_t)r * IN_CH + c];
        s += v; s2 += v * v;
    }
    atomicAdd(&sum[c], s);
    atomicAdd(&sumsq[c], s2);
}

__global__ void bn_apply_split(const float* __restrict__ x, const float* __restrict__ sum,
                               const float* __restrict__ sumsq, const float* __restrict__ g,
                               const float* __restrict__ b, bf16_t* __restrict__ hi,
                               bf16_t* __restrict__ lo, int n) {
    int i = blockIdx.x * blockDim.x + threadIdx.x;
    if (i >= n) return;
    int c = i & (IN_CH - 1);
    float mu  = sum[c]   * (1.0f / N_NODES);
    float var = sumsq[c] * (1.0f / N_NODES) - mu * mu;
    float y = (x[i] - mu) * rsqrtf(var + 1e-5f) * g[c] + b[c];
    bf16_t h = (bf16_t)y;
    hi[i] = h;
    lo[i] = (bf16_t)(y - (float)h);
}

__global__ void bn_apply_f32(const float* __restrict__ x, const float* __restrict__ sum,
                             const float* __restrict__ sumsq, const float* __restrict__ g,
                             const float* __restrict__ b, float* __restrict__ out, int n) {
    int i = blockIdx.x * blockDim.x + threadIdx.x;
    if (i >= n) return;
    int c = i & (IN_CH - 1);
    float mu  = sum[c]   * (1.0f / N_NODES);
    float var = sumsq[c] * (1.0f / N_NODES) - mu * mu;
    out[i] = (x[i] - mu) * rsqrtf(var + 1e-5f) * g[c] + b[c];
}

// ---------- Pack 256x256 weights into WMMA B-fragment order (bf16 hi/lo) ----------
// packed[(ct*8+kt)*32 + lane][j:0..15] = W[k][ct*16 + lane%16],
//   k = kt*32 + (lane<16?0:8) + (j&7) + ((j>>3)<<4)
__global__ void pack_w(const float* __restrict__ W, bf16_t* __restrict__ hi,
                       bf16_t* __restrict__ lo) {
    int i = blockIdx.x * blockDim.x + threadIdx.x;
    if (i >= COLTILES * KTILES * 32 * 16) return;
    int j    = i & 15;
    int lane = (i >> 4) & 31;
    int kt   = (i >> 9) & 7;
    int ct   = i >> 12;
    int koff = (lane < 16) ? 0 : 8;
    int k    = kt * 32 + koff + (j & 7) + ((j >> 3) << 4);
    int col  = ct * 16 + (lane & 15);
    float v = W[k * 256 + col];
    bf16_t h = (bf16_t)v;
    hi[i] = h;
    lo[i] = (bf16_t)(v - (float)h);
}

// ---------- bf16x3 WMMA GEMM: C[50000x256] = A[50000x256] * B[256x256] ----------
__device__ inline v16bf ld_a16(const bf16_t* p) {
    // 8 contiguous bf16 at p (K..K+7) and 8 at p+16 (K+16..K+23)
    v8bf x0 = *(const v8bf*)p;
    v8bf x1 = *(const v8bf*)(p + 16);
    v16bf r;
#pragma unroll
    for (int i = 0; i < 8; i++) { r[i] = x0[i]; r[i + 8] = x1[i]; }
    return r;
}

__global__ __launch_bounds__(256) void wmma_gemm_bf16x3(
    const bf16_t* __restrict__ Ahi, const bf16_t* __restrict__ Alo,
    const bf16_t* __restrict__ Bhi, const bf16_t* __restrict__ Blo,
    float* __restrict__ C) {
    int wave = (blockIdx.x * blockDim.x + threadIdx.x) >> 5;
    int lane = threadIdx.x & 31;
    if (wave >= ROWTILES * COLTILES) return;   // wave-uniform; EXEC stays all-1s
    int ct = wave / ROWTILES;                  // consecutive waves share B tiles (L2/WGP$ reuse)
    int rt = wave % ROWTILES;
    int rowA = rt * 16 + (lane & 15);
    int koff = (lane < 16) ? 0 : 8;

    v8f acc = {};
#pragma unroll
    for (int kt = 0; kt < KTILES; ++kt) {
        const bf16_t* pa  = Ahi + (size_t)rowA * 256 + kt * 32 + koff;
        const bf16_t* pal = Alo + (size_t)rowA * 256 + kt * 32 + koff;
        const bf16_t* pb  = Bhi + ((size_t)(ct * KTILES + kt) * 32 + lane) * 16;
        const bf16_t* pbl = Blo + ((size_t)(ct * KTILES + kt) * 32 + lane) * 16;
        v16bf ah = ld_a16(pa);
        v16bf al = ld_a16(pal);
        v16bf bh = *(const v16bf*)pb;
        v16bf bl = *(const v16bf*)pbl;
        // (Ahi+Alo)(Bhi+Blo) ~= AhiBhi + AhiBlo + AloBhi  (drop lo*lo ~ 2^-16 rel)
        acc = __builtin_amdgcn_wmma_f32_16x16x32_bf16(false, ah, false, bh, (short)0, acc, false, false);
        acc = __builtin_amdgcn_wmma_f32_16x16x32_bf16(false, ah, false, bl, (short)0, acc, false, false);
        acc = __builtin_amdgcn_wmma_f32_16x16x32_bf16(false, al, false, bh, (short)0, acc, false, false);
    }
    // C/D layout: VGPR r, lanes 0-15 -> M=r, lanes 16-31 -> M=r+8; N = lane%16
    int rowC = rt * 16 + ((lane >= 16) ? 8 : 0);
    int colC = ct * 16 + (lane & 15);
#pragma unroll
    for (int r = 0; r < 8; ++r)
        C[(size_t)(rowC + r) * 256 + colC] = acc[r];
}

// ---------- layer-1 attention ----------
__global__ void logits1(const float* __restrict__ h1, const float* __restrict__ asrc,
                        const float* __restrict__ adst, float* __restrict__ als,
                        float* __restrict__ ald) {
    int i = blockIdx.x * blockDim.x + threadIdx.x;  // n*4 + h
    if (i >= N_NODES * 4) return;
    int h = i & 3, n = i >> 2;
    const float* hp = h1 + (size_t)n * 256 + h * 64;
    const float* as = asrc + h * 64;
    const float* ad = adst + h * 64;
    float ss = 0.f, sd = 0.f;
#pragma unroll 4
    for (int c = 0; c < 64; c++) { float v = hp[c]; ss += v * as[c]; sd += v * ad[c]; }
    als[i] = ss; ald[i] = sd;
}

__global__ void edge_max1(const int* __restrict__ ei, const float* __restrict__ als,
                          const float* __restrict__ ald, unsigned* __restrict__ mbuf) {
    int i = blockIdx.x * blockDim.x + threadIdx.x;
    if (i >= N_EDGES * 4) return;
    int h = i & 3, e = i >> 2;
    int s = ei[e], d = ei[N_EDGES + e];
    float v = als[s * 4 + h] + ald[d * 4 + h];
    v = v > 0.f ? v : 0.2f * v;
    atomicMax(&mbuf[d * 4 + h], ford(v));
}

__global__ void edge_expsum1(const int* __restrict__ ei, const float* __restrict__ als,
                             const float* __restrict__ ald, const unsigned* __restrict__ mbuf,
                             float* __restrict__ exbuf, float* __restrict__ denom) {
    int i = blockIdx.x * blockDim.x + threadIdx.x;
    if (i >= N_EDGES * 4) return;
    int h = i & 3, e = i >> 2;
    int s = ei[e], d = ei[N_EDGES + e];
    float v = als[s * 4 + h] + ald[d * 4 + h];
    v = v > 0.f ? v : 0.2f * v;
    float ex = expf(v - funord(mbuf[d * 4 + h]));
    exbuf[i] = ex;
    atomicAdd(&denom[d * 4 + h], ex);
}

// one block per edge: 256 threads = 4 heads x 64 channels (coalesced gather + atomics)
__global__ __launch_bounds__(256) void edge_scatter1(const int* __restrict__ ei,
                                                     const float* __restrict__ exbuf,
                                                     const float* __restrict__ denom,
                                                     const float* __restrict__ h1,
                                                     float* __restrict__ acc) {
    int e = blockIdx.x;
    int t = threadIdx.x;
    int s = ei[e], d = ei[N_EDGES + e];
    int h = t >> 6;
    float alpha = exbuf[(size_t)e * 4 + h] / (denom[d * 4 + h] + 1e-16f);
    atomicAdd(&acc[(size_t)d * 256 + t], alpha * h1[(size_t)s * 256 + t]);
}

__global__ void finalize1(float* __restrict__ acc, const float* __restrict__ lb,
                          const float* __restrict__ cb, int n) {
    int i = blockIdx.x * blockDim.x + threadIdx.x;
    if (i >= n) return;
    int c = i & 255;
    float v = acc[i] + lb[c] + cb[c];
    acc[i] = v > 0.f ? v : 0.f;
}

// ---------- layer 2 (256 -> 2, one head): wave-per-node GEMV ----------
__global__ __launch_bounds__(256) void layer2_gemv(const float* __restrict__ xn2,
                                                   const float* __restrict__ W1,
                                                   const float* __restrict__ W2,
                                                   const float* __restrict__ a_s,
                                                   const float* __restrict__ a_d,
                                                   float* __restrict__ acc3,
                                                   float* __restrict__ h3,
                                                   float* __restrict__ al3s,
                                                   float* __restrict__ al3d) {
    int wave = (blockIdx.x * blockDim.x + threadIdx.x) >> 5;
    int lane = threadIdx.x & 31;
    if (wave >= N_NODES) return;
    const float* xp = xn2 + (size_t)wave * 256;
    float z0 = 0, z1 = 0, g0 = 0, g1 = 0;
    for (int k = lane; k < 256; k += 32) {
        float v = xp[k];
        z0 += v * W1[k * 2];     z1 += v * W1[k * 2 + 1];
        g0 += v * W2[k * 2];     g1 += v * W2[k * 2 + 1];
    }
#pragma unroll
    for (int off = 16; off; off >>= 1) {
        z0 += __shfl_down(z0, off, 32);
        z1 += __shfl_down(z1, off, 32);
        g0 += __shfl_down(g0, off, 32);
        g1 += __shfl_down(g1, off, 32);
    }
    if (lane == 0) {
        acc3[wave * 2] = z0; acc3[wave * 2 + 1] = z1;
        h3[wave * 2] = g0;   h3[wave * 2 + 1] = g1;
        al3s[wave] = g0 * a_s[0] + g1 * a_s[1];
        al3d[wave] = g0 * a_d[0] + g1 * a_d[1];
    }
}

__global__ void edge_max2(const int* __restrict__ ei, const float* __restrict__ als,
                          const float* __restrict__ ald, unsigned* __restrict__ m3) {
    int e = blockIdx.x * blockDim.x + threadIdx.x;
    if (e >= N_EDGES) return;
    int s = ei[e], d = ei[N_EDGES + e];
    float v = als[s] + ald[d];
    v = v > 0.f ? v : 0.2f * v;
    atomicMax(&m3[d], ford(v));
}

__global__ void edge_expsum2(const int* __restrict__ ei, const float* __restrict__ als,
                             const float* __restrict__ ald, const unsigned* __restrict__ m3,
                             float* __restrict__ ex3, float* __restrict__ den3) {
    int e = blockIdx.x * blockDim.x + threadIdx.x;
    if (e >= N_EDGES) return;
    int s = ei[e], d = ei[N_EDGES + e];
    float v = als[s] + ald[d];
    v = v > 0.f ? v : 0.2f * v;
    float ex = expf(v - funord(m3[d]));
    ex3[e] = ex;
    atomicAdd(&den3[d], ex);
}

__global__ void edge_scatter2(const int* __restrict__ ei, const float* __restrict__ ex3,
                              const float* __restrict__ den3, const float* __restrict__ h3,
                              float* __restrict__ acc3) {
    int e = blockIdx.x * blockDim.x + threadIdx.x;
    if (e >= N_EDGES) return;
    int s = ei[e], d = ei[N_EDGES + e];
    float alpha = ex3[e] / (den3[d] + 1e-16f);
    atomicAdd(&acc3[d * 2],     alpha * h3[s * 2]);
    atomicAdd(&acc3[d * 2 + 1], alpha * h3[s * 2 + 1]);
}

__global__ void finalize2(const float* __restrict__ acc3, const float* __restrict__ lb,
                          const float* __restrict__ cb, float* __restrict__ out, int n) {
    int i = blockIdx.x * blockDim.x + threadIdx.x;
    if (i >= n) return;
    int c = i & 1;
    float v = acc3[i] + lb[c] + cb[c];
    out[i] = v > 0.f ? v : 0.f;
}

// ---------- launcher ----------
extern "C" void kernel_launch(void* const* d_in, const int* in_sizes, int n_in,
                              void* d_out, int out_size, void* d_ws, size_t ws_size,
                              hipStream_t stream) {
    (void)in_sizes; (void)n_in; (void)out_size; (void)ws_size;
    const float* x       = (const float*)d_in[0];
    const int*   ei      = (const int*)  d_in[1];
    const float* bn1_g   = (const float*)d_in[2];
    const float* bn1_b   = (const float*)d_in[3];
    const float* lin1_W  = (const float*)d_in[4];
    const float* lin1_b  = (const float*)d_in[5];
    const float* con1_W  = (const float*)d_in[6];
    const float* con1_as = (const float*)d_in[7];
    const float* con1_ad = (const float*)d_in[8];
    const float* con1_b  = (const float*)d_in[9];
    const float* bn3_g   = (const float*)d_in[10];
    const float* bn3_b   = (const float*)d_in[11];
    const float* lin3_W  = (const float*)d_in[12];
    const float* lin3_b  = (const float*)d_in[13];
    const float* con3_W  = (const float*)d_in[14];
    const float* con3_as = (const float*)d_in[15];
    const float* con3_ad = (const float*)d_in[16];
    const float* con3_b  = (const float*)d_in[17];
    float* out = (float*)d_out;

    size_t off = 0;
    auto alloc = [&](size_t bytes) -> void* {
        void* p = (char*)d_ws + off;
        off += (bytes + 255) & ~(size_t)255;
        return p;
    };
    bf16_t* Ahi  = (bf16_t*)alloc(sizeof(bf16_t) * N_NODES * IN_CH);   // 25.6 MB
    bf16_t* Alo  = (bf16_t*)alloc(sizeof(bf16_t) * N_NODES * IN_CH);   // 25.6 MB
    float*  h1   = (float*) alloc(sizeof(float)  * N_NODES * IN_CH);   // 51.2 MB
    float*  acc1 = (float*) alloc(sizeof(float)  * N_NODES * IN_CH);   // 51.2 MB (z1 -> +gat -> x1)
    float*  als  = (float*) alloc(sizeof(float)  * N_NODES * 4);
    float*  ald  = (float*) alloc(sizeof(float)  * N_NODES * 4);
    unsigned* mbuf = (unsigned*)alloc(sizeof(unsigned) * N_NODES * 4);
    float*  den  = (float*) alloc(sizeof(float)  * N_NODES * 4);
    float*  exb  = (float*) alloc(sizeof(float)  * (size_t)N_EDGES * 4); // 12.8 MB
    bf16_t* WhiL = (bf16_t*)alloc(sizeof(bf16_t) * 65536);
    bf16_t* WloL = (bf16_t*)alloc(sizeof(bf16_t) * 65536);
    bf16_t* WhiC = (bf16_t*)alloc(sizeof(bf16_t) * 65536);
    bf16_t* WloC = (bf16_t*)alloc(sizeof(bf16_t) * 65536);
    float*  bns  = (float*) alloc(sizeof(float) * 512);   // sum | sumsq
    float*  h3   = (float*) alloc(sizeof(float) * N_NODES * 2);
    float*  al3s = (float*) alloc(sizeof(float) * N_NODES);
    float*  al3d = (float*) alloc(sizeof(float) * N_NODES);
    unsigned* m3 = (unsigned*)alloc(sizeof(unsigned) * N_NODES);
    float*  den3 = (float*) alloc(sizeof(float) * N_NODES);
    float*  ex3  = (float*) alloc(sizeof(float) * N_EDGES);
    float*  acc3 = (float*) alloc(sizeof(float) * N_NODES * 2);
    // xn2 (BN3 output, f32 51.2MB) overlays the dead Ahi+Alo region
    float* xn2 = (float*)Ahi;

    const unsigned NEGINF_ORD = 0x007FFFFFu;  // ford(-inf)
    const int NE = N_NODES * IN_CH;           // 12.8M

    // ---- layer 1 ----
    fill_u32<<<2, 256, 0, stream>>>((unsigned*)bns, 0u, 512);
    bn_reduce<<<512, 256, 0, stream>>>(x, bns, bns + 256, N_NODES);
    bn_apply_split<<<NE / 256, 256, 0, stream>>>(x, bns, bns + 256, bn1_g, bn1_b, Ahi, Alo, NE);
    pack_w<<<256, 256, 0, stream>>>(lin1_W, WhiL, WloL);
    pack_w<<<256, 256, 0, stream>>>(con1_W, WhiC, WloC);
    wmma_gemm_bf16x3<<<6250, 256, 0, stream>>>(Ahi, Alo, WhiL, WloL, acc1);  // z1
    wmma_gemm_bf16x3<<<6250, 256, 0, stream>>>(Ahi, Alo, WhiC, WloC, h1);    // gat hidden
    logits1<<<(N_NODES * 4 + 255) / 256, 256, 0, stream>>>(h1, con1_as, con1_ad, als, ald);
    fill_u32<<<(N_NODES * 4 + 255) / 256, 256, 0, stream>>>(mbuf, NEGINF_ORD, N_NODES * 4);
    fill_u32<<<(N_NODES * 4 + 255) / 256, 256, 0, stream>>>((unsigned*)den, 0u, N_NODES * 4);
    edge_max1<<<N_EDGES * 4 / 256, 256, 0, stream>>>(ei, als, ald, mbuf);
    edge_expsum1<<<N_EDGES * 4 / 256, 256, 0, stream>>>(ei, als, ald, mbuf, exb, den);
    edge_scatter1<<<N_EDGES, 256, 0, stream>>>(ei, exb, den, h1, acc1);
    finalize1<<<NE / 256, 256, 0, stream>>>(acc1, lin1_b, con1_b, NE);       // acc1 = x1

    // ---- layer 2 ----
    fill_u32<<<2, 256, 0, stream>>>((unsigned*)bns, 0u, 512);
    bn_reduce<<<512, 256, 0, stream>>>(acc1, bns, bns + 256, N_NODES);
    bn_apply_f32<<<NE / 256, 256, 0, stream>>>(acc1, bns, bns + 256, bn3_g, bn3_b, xn2, NE);
    layer2_gemv<<<6250, 256, 0, stream>>>(xn2, lin3_W, con3_W, con3_as, con3_ad, acc3, h3, al3s, al3d);
    fill_u32<<<(N_NODES + 255) / 256, 256, 0, stream>>>(m3, NEGINF_ORD, N_NODES);
    fill_u32<<<(N_NODES + 255) / 256, 256, 0, stream>>>((unsigned*)den3, 0u, N_NODES);
    edge_max2<<<N_EDGES / 256, 256, 0, stream>>>(ei, al3s, al3d, m3);
    edge_expsum2<<<N_EDGES / 256, 256, 0, stream>>>(ei, al3s, al3d, m3, ex3, den3);
    edge_scatter2<<<N_EDGES / 256, 256, 0, stream>>>(ei, ex3, den3, h3, acc3);
    finalize2<<<(N_NODES * 2 + 255) / 256, 256, 0, stream>>>(acc3, lin3_b, con3_b, out, N_NODES * 2);
}
